// CosformerDDiTBlock_4037269258966
// MI455X (gfx1250) — matrix operand order repaired and checked by reading, not compiled, not run on hardware
//
#include <hip/hip_runtime.h>
#include <hip/hip_bf16.h>
#include <stdint.h>
#include <stddef.h>

// Problem constants (CosformerDDiTBlock reference)
#define B_   4
#define L_   4096
#define E_   1024
#define H_   16
#define HD_  64
#define I_   4096
#define M_   (B_ * L_)          // 16384 token rows
#define SIXE (6 * E_)

// padded K-stride (elements) for LDS tiles: 40*2B = 80B -> 16B aligned rows,
// bank step 20 per row => 16 strided rows hit 16 distinct banks.
#define KPAD 40

// ---------------- WMMA types ------------------------------------------------
typedef __attribute__((ext_vector_type(16))) __bf16 v16bf;
typedef __attribute__((ext_vector_type(8)))  __bf16 v8bf;
typedef __attribute__((ext_vector_type(8)))  float  v8f;

struct bfrag { v8bf lo, hi; };

#define WMMA_BF16(a, b, c) \
  __builtin_amdgcn_wmma_f32_16x16x32_bf16(false, (a), false, (b), (short)0, (c), false, false)

__device__ __forceinline__ float siluf(float x) { return x / (1.f + __expf(-x)); }

// CDNA5 async global -> LDS copy, 16 bytes per lane (ASYNCcnt tracked).
__device__ __forceinline__ void async_b128(uint32_t lds_off, const void* gaddr) {
  asm volatile("global_load_async_to_lds_b128 %0, %1, off"
               :: "v"(lds_off), "v"(gaddr) : "memory");
}
__device__ __forceinline__ void wait_async0() {
  asm volatile("s_wait_asynccnt 0x0" ::: "memory");
}
__device__ __forceinline__ uint32_t lds_off(const void* p) {
  // AMDGPU generic pointers to LDS carry the LDS byte offset in the low 32 bits.
  return (uint32_t)(uintptr_t)p;
}

// A fragment (16x32 bf16) from an LDS tile with KPAD-element row stride.
// lane m=l&15 -> row m; lanes 0-15: K chunks [0..7],[16..23]; 16-31: [8..15],[24..31].
__device__ __forceinline__ v16bf lds_a_frag(const __bf16* tile, int row16, int lane) {
  const int m = lane & 15, hsel = lane >> 4;
  const __bf16* p = tile + (size_t)(row16 + m) * KPAD + 8 * hsel;
  bfrag t;
  t.lo = *(const v8bf*)(p);
  t.hi = *(const v8bf*)(p + 16);
  return __builtin_bit_cast(v16bf, t);
}
// B fragment (32x16 bf16, B[k][n] = W[n][k]): lane n=l&15;
// lanes 0-15: K chunks [0..7],[16..23]; 16-31: [8..15],[24..31].
__device__ __forceinline__ v16bf lds_b_frag(const __bf16* tile, int col16, int lane) {
  const int n = lane & 15, kb = (lane >> 4) * 8;
  const __bf16* p = tile + (size_t)(col16 + n) * KPAD + kb;
  bfrag t;
  t.lo = *(const v8bf*)(p);
  t.hi = *(const v8bf*)(p + 16);
  return __builtin_bit_cast(v16bf, t);
}

// ---------------- small helper kernels -------------------------------------

__global__ void k_cvt_bf16(const float* __restrict__ src, __bf16* __restrict__ dst, int n) {
  int i = blockIdx.x * blockDim.x + threadIdx.x;
  const int stride = gridDim.x * blockDim.x;
  for (; i < n; i += stride) dst[i] = (__bf16)src[i];
}

// mod[b, r] = dot(c[b,:], Wada[r,:]) + bada[r]
__global__ void __launch_bounds__(256)
k_adaln(const float* __restrict__ c, const float* __restrict__ Wada,
        const float* __restrict__ bada, float* __restrict__ mod) {
  const int o = blockIdx.x;            // 0 .. B*6E-1
  const int b = o / SIXE, r = o % SIXE;
  const float* cv = c + (size_t)b * E_;
  const float* w  = Wada + (size_t)r * E_;
  float s = 0.f;
  for (int e = threadIdx.x; e < E_; e += blockDim.x) s += cv[e] * w[e];
  __shared__ float red[256];
  red[threadIdx.x] = s; __syncthreads();
  for (int st = 128; st > 0; st >>= 1) {
    if (threadIdx.x < st) red[threadIdx.x] += red[threadIdx.x + st];
    __syncthreads();
  }
  if (threadIdx.x == 0) mod[o] = red[0] + bada[r];
}

// out_bf16[row, :] = rmsnorm(x[row,:], w) * (1 + mod[b, sc+e]) + mod[b, sh+e]
__global__ void __launch_bounds__(256)
k_norm_mod(const float* __restrict__ x, const float* __restrict__ w,
           const float* __restrict__ mod, int sh_ofs, int sc_ofs,
           __bf16* __restrict__ out) {
  const int row = blockIdx.x;          // 0..M-1
  const int b = row / L_;
  const float* xr = x + (size_t)row * E_;
  float s = 0.f;
  for (int e = threadIdx.x; e < E_; e += blockDim.x) { float v = xr[e]; s += v * v; }
  __shared__ float red[256];
  __shared__ float rinv;
  red[threadIdx.x] = s; __syncthreads();
  for (int st = 128; st > 0; st >>= 1) {
    if (threadIdx.x < st) red[threadIdx.x] += red[threadIdx.x + st];
    __syncthreads();
  }
  if (threadIdx.x == 0) rinv = rsqrtf(red[0] / (float)E_ + 1e-6f);
  __syncthreads();
  const float* mb = mod + (size_t)b * SIXE;
  for (int e = threadIdx.x; e < E_; e += blockDim.x) {
    const float xn = w[e] * xr[e] * rinv;
    out[(size_t)row * E_ + e] = (__bf16)(xn * (1.f + mb[sc_ofs + e]) + mb[sh_ofs + e]);
  }
}

// ---------------- WMMA GEMM: out[M,N] = A[M,K] @ W[N,K]^T -------------------
// 8 waves as 2(M) x 4(N); wave tile 32x64 (two A frags share each B frag).
// Block tile 64(M) x 256(N); K-step 32; async global->LDS double buffering
// (A 2x5KB + B 2x20KB = 50KB LDS). Fragments are register-prefetched so the
// 8 WMMAs per k-step issue back-to-back behind a single s_wait_dscnt.
// epi: 0 = store bf16, 1 = silu -> bf16, 2 = outf = resid + gate * acc (f32)
__global__ void __launch_bounds__(256)
k_gemm(const __bf16* __restrict__ A, const __bf16* __restrict__ W,
       int Kdim, int Ndim, int epi,
       __bf16* __restrict__ outb, float* __restrict__ outf,
       const float* __restrict__ resid, const float* __restrict__ mod, int gate_ofs) {
  __shared__ __bf16 sA[2][64 * KPAD];
  __shared__ __bf16 sB[2][256 * KPAD];
  const int t = threadIdx.x;
  const int lane = t & 31;
  const int wave = t >> 5;
  const int wm = wave & 1, wn = wave >> 1;
  const int rowBlk = blockIdx.x * 64;
  const int colBlk = blockIdx.y * 256;

  // Cooperative async fill of one k-step tile (A: 256 x 16B, B: 1024 x 16B).
  auto issue = [&](int buf, int k0) {
    {
      const int r = t >> 2, cch = t & 3;
      async_b128(lds_off(&sA[buf][r * KPAD + cch * 8]),
                 A + (size_t)(rowBlk + r) * Kdim + k0 + cch * 8);
    }
#pragma unroll
    for (int jj = 0; jj < 4; ++jj) {
      const int idx = t + jj * 256;
      const int n = idx >> 2, cch = idx & 3;
      async_b128(lds_off(&sB[buf][n * KPAD + cch * 8]),
                 W + (size_t)(colBlk + n) * Kdim + k0 + cch * 8);
    }
  };

  v8f acc[2][4] = {};
  issue(0, 0);
  wait_async0();
  __syncthreads();
  for (int k0 = 0; k0 < Kdim; k0 += 32) {
    const int cur = (k0 >> 5) & 1;
    if (k0 + 32 < Kdim) issue(cur ^ 1, k0 + 32);   // overlap DMA with compute
    // cluster all LDS fragment reads, then the WMMA chain
    const v16bf a0 = lds_a_frag(sA[cur], wm * 32, lane);
    const v16bf a1 = lds_a_frag(sA[cur], wm * 32 + 16, lane);
    v16bf bf[4];
#pragma unroll
    for (int j = 0; j < 4; ++j) bf[j] = lds_b_frag(sB[cur], wn * 64 + 16 * j, lane);
#pragma unroll
    for (int j = 0; j < 4; ++j) {
      acc[0][j] = WMMA_BF16(a0, bf[j], acc[0][j]);
      acc[1][j] = WMMA_BF16(a1, bf[j], acc[1][j]);
    }
    wait_async0();
    __syncthreads();
  }

  const int nl = lane & 15, hsel = lane >> 4;
#pragma unroll
  for (int ms = 0; ms < 2; ++ms) {
    const int row0 = rowBlk + wm * 32 + ms * 16;
    const int col0 = colBlk + wn * 64;
#pragma unroll
    for (int j = 0; j < 4; ++j) {
#pragma unroll
      for (int i = 0; i < 8; ++i) {
        const int r  = row0 + hsel * 8 + i;
        const int cc = col0 + 16 * j + nl;
        const float v = acc[ms][j][i];
        if (epi == 0) {
          outb[(size_t)r * Ndim + cc] = (__bf16)v;
        } else if (epi == 1) {
          outb[(size_t)r * Ndim + cc] = (__bf16)siluf(v);
        } else {
          const int bb = r / L_;
          const float g = mod[(size_t)bb * SIXE + gate_ofs + cc];
          outf[(size_t)r * Ndim + cc] = resid[(size_t)r * Ndim + cc] + g * v;
        }
      }
    }
  }
}

// Dual-B GEMM: gu = silu(A@Wg^T) * (A@Wu^T), bf16 out. N = I, K = E.
// 8 waves -> 64(M) x 128(N) block tile (fits two B tiles: 51KB LDS).
__global__ void __launch_bounds__(256)
k_gemm_gu(const __bf16* __restrict__ A, const __bf16* __restrict__ Wg,
          const __bf16* __restrict__ Wu, __bf16* __restrict__ out) {
  __shared__ __bf16 sA[2][64 * KPAD];
  __shared__ __bf16 sG[2][128 * KPAD];
  __shared__ __bf16 sU[2][128 * KPAD];
  const int t = threadIdx.x;
  const int lane = t & 31;
  const int wave = t >> 5;
  const int wm = wave & 3, wn = wave >> 2;
  const int rowBlk = blockIdx.x * 64;
  const int colBlk = blockIdx.y * 128;

  // A: 256 x 16B (1/thread); G,U: 512 x 16B each (2/thread each).
  auto issue = [&](int buf, int k0) {
    {
      const int r = t >> 2, cch = t & 3;
      async_b128(lds_off(&sA[buf][r * KPAD + cch * 8]),
                 A + (size_t)(rowBlk + r) * E_ + k0 + cch * 8);
    }
#pragma unroll
    for (int jj = 0; jj < 2; ++jj) {
      const int idx = t + jj * 256;
      const int n = idx >> 2, cch = idx & 3;
      async_b128(lds_off(&sG[buf][n * KPAD + cch * 8]),
                 Wg + (size_t)(colBlk + n) * E_ + k0 + cch * 8);
      async_b128(lds_off(&sU[buf][n * KPAD + cch * 8]),
                 Wu + (size_t)(colBlk + n) * E_ + k0 + cch * 8);
    }
  };

  v8f ag[4] = {}, au[4] = {};
  issue(0, 0);
  wait_async0();
  __syncthreads();
  for (int k0 = 0; k0 < E_; k0 += 32) {
    const int cur = (k0 >> 5) & 1;
    if (k0 + 32 < E_) issue(cur ^ 1, k0 + 32);
    const v16bf a = lds_a_frag(sA[cur], wm * 16, lane);
    v16bf bg[4], bu[4];
#pragma unroll
    for (int j = 0; j < 4; ++j) {
      bg[j] = lds_b_frag(sG[cur], wn * 64 + 16 * j, lane);
      bu[j] = lds_b_frag(sU[cur], wn * 64 + 16 * j, lane);
    }
#pragma unroll
    for (int j = 0; j < 4; ++j) {
      ag[j] = WMMA_BF16(a, bg[j], ag[j]);
      au[j] = WMMA_BF16(a, bu[j], au[j]);
    }
    wait_async0();
    __syncthreads();
  }

  const int row0 = rowBlk + wm * 16, col0 = colBlk + wn * 64;
  const int nl = lane & 15, hsel = lane >> 4;
#pragma unroll
  for (int j = 0; j < 4; ++j) {
#pragma unroll
    for (int i = 0; i < 8; ++i) {
      const int r  = row0 + hsel * 8 + i;
      const int cc = col0 + 16 * j + nl;
      out[(size_t)r * I_ + cc] = (__bf16)(siluf(ag[j][i]) * au[j][i]);
    }
  }
}

// ---------------- RoPE + cosFormer feature map ------------------------------
// src: (B,L,E) bf16 (post silu for q/k). dst: (B,H,L,2*HD) bf16.
__global__ void __launch_bounds__(256)
k_ropecos(const __bf16* __restrict__ src, const float* __restrict__ cosT,
          const float* __restrict__ sinT, __bf16* __restrict__ dst) {
  const int tid = blockIdx.x * blockDim.x + threadIdx.x;   // B*L*H*HD = 2^24
  const int d  = tid & (HD_ - 1);
  const int hh = (tid >> 6) & (H_ - 1);
  const int l  = (tid >> 10) & (L_ - 1);
  const int b  = tid >> 22;
  const size_t base = ((size_t)(b * L_ + l)) * E_ + hh * HD_;
  const float q  = (float)src[base + d];
  const float qo = (float)src[base + (d ^ 32)];
  const float rot = (d < 32) ? -qo : qo;
  const float qr = q * cosT[l * HD_ + d] + rot * sinT[l * HD_ + d];
  const float idx = 1.5707963267948966f * (float)(l + 1) / (float)L_;
  const float sw = __sinf(idx), cw = __cosf(idx);
  const size_t ob = ((size_t)(b * H_ + hh) * L_ + l) * (2 * HD_);
  dst[ob + d]       = (__bf16)(qr * sw);
  dst[ob + HD_ + d] = (__bf16)(qr * cw);
}

// ---------------- kv = K~^T V per (b,h); ksum = sum_l K~[l] -----------------
__global__ void __launch_bounds__(256)
k_kv(const __bf16* __restrict__ kf, const __bf16* __restrict__ v,
     float* __restrict__ kv, float* __restrict__ ksum) {
  const int bh = blockIdx.x;                 // 0..B*H-1
  const int b = bh / H_, hh = bh % H_;
  const __bf16* krow = kf + (size_t)bh * L_ * (2 * HD_);
  const int t = threadIdx.x;
  const int dv = t & 63, dk0 = (t >> 6) * 32;
  float acc[32];
#pragma unroll
  for (int i = 0; i < 32; ++i) acc[i] = 0.f;
  float ks_acc = 0.f;
  __shared__ float klds[128];
  for (int l = 0; l < L_; ++l) {
    if (t < 128) klds[t] = (float)krow[(size_t)l * 128 + t];
    __syncthreads();
    const float vv = (float)v[((size_t)(b * L_ + l)) * E_ + hh * HD_ + dv];
#pragma unroll
    for (int i = 0; i < 32; ++i) acc[i] += klds[dk0 + i] * vv;
    if (t < 128) ks_acc += klds[t];
    __syncthreads();
  }
  float* kvb = kv + (size_t)bh * 128 * 64;
#pragma unroll
  for (int i = 0; i < 32; ++i) kvb[(size_t)(dk0 + i) * 64 + dv] = acc[i];
  if (t < 128) ksum[bh * 128 + t] = ks_acc;
}

// ---------------- attn[l] = (q~[l] . kv) / max(q~[l] . ksum, eps) -----------
__global__ void __launch_bounds__(256)
k_attn(const __bf16* __restrict__ qf, const float* __restrict__ kv,
       const float* __restrict__ ksum, __bf16* __restrict__ attn) {
  const int bh = blockIdx.y;
  const int b = bh / H_, hh = bh % H_;
  const int l0 = blockIdx.x * 4;
  __shared__ float kvs[128 * 64];
  __shared__ float kss[128];
  __shared__ float qsl[4 * 128];
  const int t = threadIdx.x;
  const float* kvb = kv + (size_t)bh * 128 * 64;
  for (int i = t; i < 128 * 64; i += 256) kvs[i] = kvb[i];
  if (t < 128) kss[t] = ksum[bh * 128 + t];
  const __bf16* qb = qf + ((size_t)bh * L_ + l0) * 128;
  for (int i = t; i < 512; i += 256) qsl[i] = (float)qb[i];
  __syncthreads();
  const int lq = t >> 6, dv = t & 63;
  const float* q = qsl + lq * 128;
  float zden = 0.f, o = 0.f;
#pragma unroll 8
  for (int dk = 0; dk < 128; ++dk) {
    const float qv = q[dk];
    zden += qv * kss[dk];
    o    += qv * kvs[dk * 64 + dv];
  }
  const float z = 1.f / fmaxf(zden, 1e-6f);
  attn[((size_t)(b * L_ + l0 + lq)) * E_ + hh * HD_ + dv] = (__bf16)(o * z);
}

// ---------------- host launch ----------------------------------------------
extern "C" void kernel_launch(void* const* d_in, const int* in_sizes, int n_in,
                              void* d_out, int out_size, void* d_ws, size_t ws_size,
                              hipStream_t stream) {
  const float* h    = (const float*)d_in[0];
  const float* c    = (const float*)d_in[1];
  const float* cosT = (const float*)d_in[2];
  const float* sinT = (const float*)d_in[3];
  const float* Wq   = (const float*)d_in[4];
  const float* Wk   = (const float*)d_in[5];
  const float* Wv   = (const float*)d_in[6];
  const float* Wo   = (const float*)d_in[7];
  const float* Wg   = (const float*)d_in[8];
  const float* Wu   = (const float*)d_in[9];
  const float* Wd   = (const float*)d_in[10];
  const float* w1   = (const float*)d_in[11];
  const float* w2   = (const float*)d_in[12];
  const float* Wada = (const float*)d_in[13];
  const float* bada = (const float*)d_in[14];
  float* out = (float*)d_out;

  char* ws = (char*)d_ws;
  size_t off = 0;
  auto alloc = [&](size_t bytes) -> char* {
    char* p = ws + off;
    off += (bytes + 255) & ~(size_t)255;
    return p;
  };

  __bf16* wq_b = (__bf16*)alloc((size_t)E_ * E_ * 2);
  __bf16* wk_b = (__bf16*)alloc((size_t)E_ * E_ * 2);
  __bf16* wv_b = (__bf16*)alloc((size_t)E_ * E_ * 2);
  __bf16* wo_b = (__bf16*)alloc((size_t)E_ * E_ * 2);
  __bf16* wg_b = (__bf16*)alloc((size_t)I_ * E_ * 2);
  __bf16* wu_b = (__bf16*)alloc((size_t)I_ * E_ * 2);
  __bf16* wd_b = (__bf16*)alloc((size_t)E_ * I_ * 2);
  float*  mod  = (float*)alloc((size_t)B_ * SIXE * 4);
  __bf16* x1bf = (__bf16*)alloc((size_t)M_ * E_ * 2);      // reused as x2
  __bf16* qs_b = (__bf16*)alloc((size_t)M_ * E_ * 2);      // reused as attn out
  __bf16* ks_b = (__bf16*)alloc((size_t)M_ * E_ * 2);
  __bf16* vs_b = (__bf16*)alloc((size_t)M_ * E_ * 2);
  __bf16* qf   = (__bf16*)alloc((size_t)M_ * 2 * E_ * 2);  // (B,H,L,128)
  __bf16* kf   = (__bf16*)alloc((size_t)M_ * 2 * E_ * 2);  // (B,H,L,128)
  float*  kvb  = (float*)alloc((size_t)B_ * H_ * 128 * 64 * 4);
  float*  ksum = (float*)alloc((size_t)B_ * H_ * 128 * 4);
  float*  y1   = (float*)alloc((size_t)M_ * E_ * 4);       // h + g1*attn
  __bf16* gu   = qf;        // qf+kf span M_*I_ bf16, dead after attention
  __bf16* attnb = qs_b;     // qs_b dead after RoPE

  const int T = 256;

  // 1) weights -> bf16
  k_cvt_bf16<<<2048, T, 0, stream>>>(Wq, wq_b, E_ * E_);
  k_cvt_bf16<<<2048, T, 0, stream>>>(Wk, wk_b, E_ * E_);
  k_cvt_bf16<<<2048, T, 0, stream>>>(Wv, wv_b, E_ * E_);
  k_cvt_bf16<<<2048, T, 0, stream>>>(Wo, wo_b, E_ * E_);
  k_cvt_bf16<<<4096, T, 0, stream>>>(Wg, wg_b, I_ * E_);
  k_cvt_bf16<<<4096, T, 0, stream>>>(Wu, wu_b, I_ * E_);
  k_cvt_bf16<<<4096, T, 0, stream>>>(Wd, wd_b, E_ * I_);

  // 2) adaLN modulation
  k_adaln<<<B_ * SIXE, T, 0, stream>>>(c, Wada, bada, mod);

  // 3) norm1 + modulate  (sh1 @0, sc1 @E)
  k_norm_mod<<<M_, T, 0, stream>>>(h, w1, mod, 0, E_, x1bf);

  // 4) QKV projections (WMMA); silu on Q/K
  dim3 gE(M_ / 64, E_ / 256);
  k_gemm<<<gE, T, 0, stream>>>(x1bf, wq_b, E_, E_, 1, qs_b, nullptr, nullptr, nullptr, 0);
  k_gemm<<<gE, T, 0, stream>>>(x1bf, wk_b, E_, E_, 1, ks_b, nullptr, nullptr, nullptr, 0);
  k_gemm<<<gE, T, 0, stream>>>(x1bf, wv_b, E_, E_, 0, vs_b, nullptr, nullptr, nullptr, 0);

  // 5) RoPE + cosFormer feature map
  const int nrope = B_ * L_ * H_ * HD_;
  k_ropecos<<<nrope / T, T, 0, stream>>>(qs_b, cosT, sinT, qf);
  k_ropecos<<<nrope / T, T, 0, stream>>>(ks_b, cosT, sinT, kf);

  // 6) kv / ksum per (b,h); then O(L) attention
  k_kv<<<B_ * H_, T, 0, stream>>>(kf, vs_b, kvb, ksum);
  dim3 gA(L_ / 4, B_ * H_);
  k_attn<<<gA, T, 0, stream>>>(qf, kvb, ksum, attnb);

  // 7) Wo projection fused with residual + gate_msa (@2E): y1 = h + g1*acc
  k_gemm<<<gE, T, 0, stream>>>(attnb, wo_b, E_, E_, 2, nullptr, y1, h, mod, 2 * E_);

  // 8) norm2 + modulate (sh2 @3E, sc2 @4E)
  k_norm_mod<<<M_, T, 0, stream>>>(y1, w2, mod, 3 * E_, 4 * E_, x1bf);

  // 9) fused gate/up GEMM: gu = silu(x@Wg^T) * (x@Wu^T)
  dim3 gI(M_ / 64, I_ / 128);
  k_gemm_gu<<<gI, T, 0, stream>>>(x1bf, wg_b, wu_b, gu);

  // 10) down projection fused with residual + gate_mlp (@5E): out = y1 + g2*acc
  k_gemm<<<gE, T, 0, stream>>>(gu, wd_b, I_, E_, 2, nullptr, out, y1, mod, 5 * E_);

  (void)in_sizes; (void)n_in; (void)out_size; (void)ws_size;
}
